// CustomGINE_81758997447423
// MI455X (gfx1250) — compile-verified, matrix operand order checked by reading, not run on hardware
//
#include <hip/hip_runtime.h>
#include <hip/hip_bf16.h>

typedef __attribute__((ext_vector_type(16))) _Float16 v16h;
typedef __attribute__((ext_vector_type(8)))  float    v8f;

#define N_NODES   100000
#define N_EDGES   1600000
#define DIM       64
#define BN_EPS    1e-5f
#define N_TILES   (N_NODES / 16)          // 6250 exact 16-row WMMA tiles
#define AGG_FLOATS (N_NODES * DIM)        // 6,400,000 floats = 25.6 MB
// ws layout (floats / bytes):
//   [0, AGG_FLOATS)            agg, later reused in-place as t = h@W1+b1
//   colsum[64], colsq[64], bn_a[64], bn_b[64]   (256 floats)
//   then W1 packed f16 fragments (4096 halves), W2 packed (4096 halves)

// K-index a lane's i-th half occupies in a 16-bit 16x32 WMMA fragment
// (ISA 7.12.2: lanes 0-15 -> K 0..7,16..23 ; lanes 16-31 -> K 8..15,24..31)
__device__ __forceinline__ int k_of(int g, int i) {
    return (i < 8) ? (i + g * 8) : (16 + (i - 8) + g * 8);
}

// ---------------- zero workspace (agg + stat accumulators) ----------------
__global__ __launch_bounds__(256) void zero_ws(float* p, int n) {
    int i = blockIdx.x * 256 + threadIdx.x;
    if (i < n) p[i] = 0.0f;
}

// ---------------- pack W1/W2 (f32 row-major 64x64) into f16 B-fragments ---
__global__ __launch_bounds__(256) void pack_weights(const float* __restrict__ W1,
                                                    const float* __restrict__ W2,
                                                    _Float16* __restrict__ P1,
                                                    _Float16* __restrict__ P2) {
    int f    = threadIdx.x >> 5;          // fragment id 0..7  (ntile*2 + kstep)
    int lane = threadIdx.x & 31;
    int nt   = f >> 1, ks = f & 1;
    int n    = nt * 16 + (lane & 15);
    int g    = (lane >> 4) & 1;
    for (int m = 0; m < 2; ++m) {
        const float* W = m ? W2 : W1;
        _Float16*    P = m ? P2 : P1;
        _Float16* dst = P + (f * 32 + lane) * 16;
        #pragma unroll
        for (int i = 0; i < 16; ++i) {
            int k = ks * 32 + k_of(g, i);
            dst[i] = (_Float16)W[k * DIM + n];
        }
    }
}

// ---------------- edge phase: msg = relu(x[src]+emb[attr]); scatter-add ---
__global__ __launch_bounds__(256) void edge_scatter(const float* __restrict__ x,
                                                    const float* __restrict__ emb,
                                                    const int* __restrict__ src,
                                                    const int* __restrict__ dst,
                                                    const int* __restrict__ attr,
                                                    float* __restrict__ agg) {
    int gid = blockIdx.x * 256 + threadIdx.x;   // E*16 work items (4 floats each)
    int e = gid >> 4;
    if (e >= N_EDGES) return;
    int c = (gid & 15) << 2;
    int s = src[e], d = dst[e], a = attr[e];
    const float4 xv = *(const float4*)(x   + s * DIM + c);
    const float4 ev = *(const float4*)(emb + a * DIM + c);
    float m0 = fmaxf(xv.x + ev.x, 0.0f);
    float m1 = fmaxf(xv.y + ev.y, 0.0f);
    float m2 = fmaxf(xv.z + ev.z, 0.0f);
    float m3 = fmaxf(xv.w + ev.w, 0.0f);
    float* o = agg + d * DIM + c;
    atomicAdd(o + 0, m0);
    atomicAdd(o + 1, m1);
    atomicAdd(o + 2, m2);
    atomicAdd(o + 3, m3);
}

// ---------------- GEMM1: t = ((1+eps)x + agg)@W1 + b1, + BN stat atomics --
__global__ __launch_bounds__(256) void gemm1_stats(const float* __restrict__ x,
                                                   const float* __restrict__ epsp,
                                                   const float* __restrict__ b1,
                                                   float* __restrict__ tbuf, // agg in, t out (in-place)
                                                   const _Float16* __restrict__ w1p,
                                                   float* __restrict__ colsum,
                                                   float* __restrict__ colsq) {
    int wave = blockIdx.x * 8 + (threadIdx.x >> 5);
    if (wave >= N_TILES) return;                  // whole-wave skip: EXEC stays all-1 inside
    int lane = threadIdx.x & 31;
    int g    = (lane >> 4) & 1;
    int arow = wave * 16 + (lane & 15);
    float epsv = 1.0f + epsp[0];

    const float* xr = x    + arow * DIM;
    const float* gr = tbuf + arow * DIM;

    v16h afrag[2];
    #pragma unroll
    for (int ks = 0; ks < 2; ++ks) {
        int b0 = ks * 32 + g * 8;                 // two contiguous 8-float K chunks
        int b1o = b0 + 16;
        float4 xa = *(const float4*)(xr + b0);
        float4 xb = *(const float4*)(xr + b0 + 4);
        float4 xc = *(const float4*)(xr + b1o);
        float4 xd = *(const float4*)(xr + b1o + 4);
        float4 ga = *(const float4*)(gr + b0);
        float4 gb = *(const float4*)(gr + b0 + 4);
        float4 gc = *(const float4*)(gr + b1o);
        float4 gd = *(const float4*)(gr + b1o + 4);
        afrag[ks][0]  = (_Float16)(epsv * xa.x + ga.x);
        afrag[ks][1]  = (_Float16)(epsv * xa.y + ga.y);
        afrag[ks][2]  = (_Float16)(epsv * xa.z + ga.z);
        afrag[ks][3]  = (_Float16)(epsv * xa.w + ga.w);
        afrag[ks][4]  = (_Float16)(epsv * xb.x + gb.x);
        afrag[ks][5]  = (_Float16)(epsv * xb.y + gb.y);
        afrag[ks][6]  = (_Float16)(epsv * xb.z + gb.z);
        afrag[ks][7]  = (_Float16)(epsv * xb.w + gb.w);
        afrag[ks][8]  = (_Float16)(epsv * xc.x + gc.x);
        afrag[ks][9]  = (_Float16)(epsv * xc.y + gc.y);
        afrag[ks][10] = (_Float16)(epsv * xc.z + gc.z);
        afrag[ks][11] = (_Float16)(epsv * xc.w + gc.w);
        afrag[ks][12] = (_Float16)(epsv * xd.x + gd.x);
        afrag[ks][13] = (_Float16)(epsv * xd.y + gd.y);
        afrag[ks][14] = (_Float16)(epsv * xd.z + gd.z);
        afrag[ks][15] = (_Float16)(epsv * xd.w + gd.w);
    }

    v8f acc[4];
    #pragma unroll
    for (int nt = 0; nt < 4; ++nt) {
        acc[nt] = (v8f){0.f, 0.f, 0.f, 0.f, 0.f, 0.f, 0.f, 0.f};
        #pragma unroll
        for (int ks = 0; ks < 2; ++ks) {
            v16h bfrag = *(const v16h*)(w1p + ((nt * 2 + ks) * 32 + lane) * 16);
            acc[nt] = __builtin_amdgcn_wmma_f32_16x16x32_f16(
                false, afrag[ks], false, bfrag, (short)0, acc[nt], false, false);
        }
    }

    // bias, in-place store of t over agg, per-column sum / sum^2 for BN
    #pragma unroll
    for (int nt = 0; nt < 4; ++nt) {
        int col = nt * 16 + (lane & 15);
        float bv = b1[col];
        float s = 0.f, sq = 0.f;
        #pragma unroll
        for (int r = 0; r < 8; ++r) {
            float v = acc[nt][r] + bv;
            int m = r + g * 8;
            tbuf[(wave * 16 + m) * DIM + col] = v;
            s += v; sq += v * v;
        }
        atomicAdd(&colsum[col], s);
        atomicAdd(&colsq[col], sq);
    }
}

// ---------------- fold BN stats into per-column affine: y = t*a + b -------
__global__ __launch_bounds__(64) void bn_finalize(const float* __restrict__ colsum,
                                                  const float* __restrict__ colsq,
                                                  const float* __restrict__ gamma,
                                                  const float* __restrict__ beta,
                                                  float* __restrict__ bn_a,
                                                  float* __restrict__ bn_b) {
    int c = threadIdx.x;
    float invn = 1.0f / (float)N_NODES;
    float mu   = colsum[c] * invn;
    float var  = colsq[c] * invn - mu * mu;
    float a    = gamma[c] * rsqrtf(var + BN_EPS);
    bn_a[c] = a;
    bn_b[c] = beta[c] - mu * a;
}

// ---------------- GEMM2: out = relu(t*a+b) @ W2 + b2 ----------------------
__global__ __launch_bounds__(256) void gemm2_out(const float* __restrict__ tbuf,
                                                 const float* __restrict__ bn_a,
                                                 const float* __restrict__ bn_b,
                                                 const float* __restrict__ b2,
                                                 const _Float16* __restrict__ w2p,
                                                 float* __restrict__ out) {
    __shared__ float sa[DIM];
    __shared__ float sb[DIM];
    if (threadIdx.x < DIM) {
        sa[threadIdx.x] = bn_a[threadIdx.x];
        sb[threadIdx.x] = bn_b[threadIdx.x];
    }
    __syncthreads();

    int wave = blockIdx.x * 8 + (threadIdx.x >> 5);
    if (wave >= N_TILES) return;
    int lane = threadIdx.x & 31;
    int g    = (lane >> 4) & 1;
    int arow = wave * 16 + (lane & 15);
    const float* tr = tbuf + arow * DIM;

    v16h afrag[2];
    #pragma unroll
    for (int ks = 0; ks < 2; ++ks) {
        int b0  = ks * 32 + g * 8;
        int b1o = b0 + 16;
        float4 ta = *(const float4*)(tr + b0);
        float4 tb = *(const float4*)(tr + b0 + 4);
        float4 tc = *(const float4*)(tr + b1o);
        float4 td = *(const float4*)(tr + b1o + 4);
        float v[16] = {ta.x, ta.y, ta.z, ta.w, tb.x, tb.y, tb.z, tb.w,
                       tc.x, tc.y, tc.z, tc.w, td.x, td.y, td.z, td.w};
        #pragma unroll
        for (int i = 0; i < 16; ++i) {
            int k = (i < 8) ? (b0 + i) : (b1o + i - 8);
            float y = fmaxf(v[i] * sa[k] + sb[k], 0.0f);   // BN + ReLU fused
            afrag[ks][i] = (_Float16)y;
        }
    }

    v8f acc[4];
    #pragma unroll
    for (int nt = 0; nt < 4; ++nt) {
        acc[nt] = (v8f){0.f, 0.f, 0.f, 0.f, 0.f, 0.f, 0.f, 0.f};
        #pragma unroll
        for (int ks = 0; ks < 2; ++ks) {
            v16h bfrag = *(const v16h*)(w2p + ((nt * 2 + ks) * 32 + lane) * 16);
            acc[nt] = __builtin_amdgcn_wmma_f32_16x16x32_f16(
                false, afrag[ks], false, bfrag, (short)0, acc[nt], false, false);
        }
    }

    #pragma unroll
    for (int nt = 0; nt < 4; ++nt) {
        int col = nt * 16 + (lane & 15);
        float bv = b2[col];
        #pragma unroll
        for (int r = 0; r < 8; ++r) {
            int m = r + g * 8;
            out[(wave * 16 + m) * DIM + col] = acc[nt][r] + bv;
        }
    }
}

extern "C" void kernel_launch(void* const* d_in, const int* in_sizes, int n_in,
                              void* d_out, int out_size, void* d_ws, size_t ws_size,
                              hipStream_t stream) {
    const float* x     = (const float*)d_in[0];
    const float* emb   = (const float*)d_in[1];
    const float* eps   = (const float*)d_in[2];
    const float* W1    = (const float*)d_in[3];
    const float* b1    = (const float*)d_in[4];
    const float* gamma = (const float*)d_in[5];
    const float* beta  = (const float*)d_in[6];
    const float* W2    = (const float*)d_in[7];
    const float* b2    = (const float*)d_in[8];
    const int*   eidx  = (const int*)d_in[9];   // (2, E): [0..E)=src, [E..2E)=dst
    const int*   eattr = (const int*)d_in[10];
    float* out = (float*)d_out;

    float* agg    = (float*)d_ws;               // also t-buffer (in-place)
    float* colsum = agg + AGG_FLOATS;
    float* colsq  = colsum + 64;
    float* bn_a   = colsum + 128;
    float* bn_b   = colsum + 192;
    _Float16* w1p = (_Float16*)(colsum + 256);
    _Float16* w2p = w1p + 4096;

    // 1. zero agg + stat accumulators (deterministic per call)
    int nzero = AGG_FLOATS + 128;
    zero_ws<<<(nzero + 255) / 256, 256, 0, stream>>>(agg, nzero);
    // 2. pack weights into wave32 WMMA f16 B-fragment layout
    pack_weights<<<1, 256, 0, stream>>>(W1, W2, w1p, w2p);
    // 3. edge message + scatter-sum (L2-resident atomics)
    edge_scatter<<<(N_EDGES * 16) / 256, 256, 0, stream>>>(
        x, emb, eidx, eidx + N_EDGES, eattr, agg);
    // 4. GEMM1 + bias + BN statistics
    int gblocks = (N_TILES + 7) / 8;
    gemm1_stats<<<gblocks, 256, 0, stream>>>(x, eps, b1, agg, w1p, colsum, colsq);
    // 5. fold stats into per-column affine
    bn_finalize<<<1, 64, 0, stream>>>(colsum, colsq, gamma, beta, bn_a, bn_b);
    // 6. BN + ReLU + GEMM2 + bias -> out
    gemm2_out<<<gblocks, 256, 0, stream>>>(agg, bn_a, bn_b, b2, w2p, out);
}
